// DecoderNetwork_39419209843029
// MI455X (gfx1250) — matrix operand order repaired and checked by reading, not compiled, not run on hardware
//
#include <hip/hip_runtime.h>
#include <hip/hip_bf16.h>

// ---------------------------------------------------------------------------
// DecoderNetwork (teacher-forced 2-layer LSTM + policy/critic heads), gfx1250.
// All matrix math in bf16 WMMA (v_wmma_f32_16x16x32_bf16) with f32 accumulate.
// ---------------------------------------------------------------------------

typedef __bf16 bf16_t;
typedef __attribute__((ext_vector_type(16))) __bf16 v16bf;
typedef __attribute__((ext_vector_type(8)))  __bf16 v8bf;
typedef __attribute__((ext_vector_type(8)))  float  v8f;

#define Bn 256      // batch
#define Tn 128      // timesteps
#define Hn 1024     // hidden
#define On 512      // output vocab
#define Ln 2        // LSTM layers
#define TBn (Tn * Bn)

// ------------------------------ WMMA helpers -------------------------------

__device__ __forceinline__ v8f wmma_bf16(v16bf a, v16bf b, v8f c) {
  // (neg_a, A, neg_b, B, c_mod, C, reuse_a, reuse_b)
  return __builtin_amdgcn_wmma_f32_16x16x32_bf16(false, a, false, b, (short)0, c,
                                                 false, false);
}

// A operand: 16x32 (MxK) bf16, row-major source with leading dim lda.
// Layout: lanes 0-15 hold M=lane, K=0..15 (elems 0-15);
// lanes 16-31 hold M=lane-16, K=16..31.
__device__ __forceinline__ v16bf load_a_tile(const bf16_t* __restrict__ A, int lda,
                                             int m0, int k0, int lane) {
  const int row = m0 + (lane & 15);
  const int kh  = (lane >> 4) << 4;               // 0 or 16
  const bf16_t* p = A + (size_t)row * lda + k0 + kh;
  union { v16bf v; v8bf h[2]; } u;
  u.h[0] = *(const v8bf*)(p);                     // K = k0+kh .. +7
  u.h[1] = *(const v8bf*)(p + 8);                 // K = k0+kh+8 .. +15
  return u.v;
}

// B operand: 32x16 (KxN). B(k,n) = W[n][k] with W row-major (ld = ldw), i.e.
// lane n holds column n = one contiguous 16-element slice of W row n.
// lanes 0-15: K=0..15, lanes 16-31: K=16..31.
__device__ __forceinline__ v16bf load_b_tile(const bf16_t* __restrict__ W, int ldw,
                                             int n0, int k0, int lane) {
  const int col  = n0 + (lane & 15);
  const int koff = (lane >> 4) << 4;              // 0 or 16
  return *(const v16bf*)(W + (size_t)col * ldw + k0 + koff);
}

__device__ __forceinline__ float sigmf(float x) { return 1.0f / (1.0f + __expf(-x)); }

__device__ __forceinline__ float wave_rmax(float v) {
#pragma unroll
  for (int off = 16; off; off >>= 1) v = fmaxf(v, __shfl_xor(v, off, 32));
  return v;
}
__device__ __forceinline__ float wave_rsum(float v) {
#pragma unroll
  for (int off = 16; off; off >>= 1) v += __shfl_xor(v, off, 32);
  return v;
}

// --------------------------- one-time converters ---------------------------

// Wcat[l][r][0:H] = W_ih[l][r][:], Wcat[l][r][H:2H] = W_hh[l][r][:]  (bf16)
__global__ void wcat_convert_kernel(const float* __restrict__ Wih,
                                    const float* __restrict__ Whh,
                                    bf16_t* __restrict__ Wcat) {
  size_t idx = (size_t)blockIdx.x * blockDim.x + threadIdx.x; // L*4H*2H threads
  size_t c  = idx % (2 * Hn);
  size_t rl = idx / (2 * Hn);                                 // l*4H + r
  float v = (c < Hn) ? Wih[rl * Hn + c] : Whh[rl * Hn + (c - Hn)];
  Wcat[idx] = (bf16_t)v;
}

__global__ void f2bf_kernel(const float* __restrict__ src, bf16_t* __restrict__ dst,
                            int n) {
  int i = blockIdx.x * blockDim.x + threadIdx.x;
  if (i < n) dst[i] = (bf16_t)src[i];
}

__global__ void addf_kernel(const float* __restrict__ a, const float* __restrict__ b,
                            float* __restrict__ dst, int n) {
  int i = blockIdx.x * blockDim.x + threadIdx.x;
  if (i < n) dst[i] = a[i] + b[i];
}

__global__ void copyf_kernel(const float* __restrict__ src, float* __restrict__ dst,
                             int n) {
  int i = blockIdx.x * blockDim.x + threadIdx.x;
  if (i < n) dst[i] = src[i];
}

// WcT[o][p] = W_c[p][o]  (bf16) -- so B-operand columns of (pi @ W_c) are contiguous
__global__ void wcT_kernel(const float* __restrict__ Wc, bf16_t* __restrict__ WcT) {
  int idx = blockIdx.x * blockDim.x + threadIdx.x;    // On*On
  int o = idx / On, p = idx % On;
  WcT[idx] = (bf16_t)Wc[p * On + o];
}

// h-state init: A0[b][H+j] = h0[0][b][j]; A1[b][H+j] = h0[1][b][j]
__global__ void inith_kernel(const float* __restrict__ h0, bf16_t* __restrict__ A0,
                             bf16_t* __restrict__ A1) {
  int idx = blockIdx.x * blockDim.x + threadIdx.x;    // L*B*H
  int l = idx / (Bn * Hn);
  int b = (idx / Hn) % Bn;
  int j = idx % Hn;
  bf16_t v = (bf16_t)h0[idx];
  bf16_t* dst = (l == 0) ? A0 : A1;
  dst[(size_t)b * (2 * Hn) + Hn + j] = v;
}

// ------------------------------- per-step ----------------------------------

// A0x[b][0:H] = embedding_bf16[token], token = (t==0) ? 0 : actions[b, t-1]
__global__ void embed_kernel(const bf16_t* __restrict__ emb,
                             const int* __restrict__ actions,
                             bf16_t* __restrict__ A0x, int t) {
  int idx = blockIdx.x * blockDim.x + threadIdx.x;    // B*H/8
  int b  = idx / (Hn / 8);
  int j8 = idx % (Hn / 8);
  int tok = (t == 0) ? 0 : actions[b * Tn + (t - 1)];
  const v8bf* src = (const v8bf*)(emb + (size_t)tok * Hn) + j8;
  v8bf* dst = (v8bf*)(A0x + (size_t)b * (2 * Hn)) + j8;
  *dst = *src;
}

// One fused LSTM layer step: gates = A @ Wcatᵀ + bias, cell update, h fanout.
// Each wave: 4 gate tiles (i,f,g,o) of one 16(batch)x16(hidden) block; the A
// operand is shared across the 4 WMMAs per k-iteration.
__global__ __launch_bounds__(128) void lstm_step_kernel(
    const bf16_t* __restrict__ A,      // [B][2H]  (x | h_prev) bf16
    const bf16_t* __restrict__ W,      // [4H][2H] Wcat bf16 (gate-major rows)
    const float*  __restrict__ bias,   // [4H]     b_ih + b_hh
    float*        __restrict__ cstate, // [B][H]   in/out (in-place safe)
    bf16_t* __restrict__ hdst0, int ld0,
    bf16_t* __restrict__ hdst1, int ld1) {
  const int lane = threadIdx.x & 31;
  const int wave = threadIdx.x >> 5;
  const int m0 = blockIdx.x << 4;                      // batch tile
  const int n0 = (blockIdx.y * 4 + wave) << 4;         // hidden tile
  const int lda = 2 * Hn;

  v8f ai = {}, af = {}, ag = {}, ao = {};
  for (int k0 = 0; k0 < 2 * Hn; k0 += 32) {
    v16bf a  = load_a_tile(A, lda, m0, k0, lane);
    v16bf wi = load_b_tile(W, lda, 0 * Hn + n0, k0, lane);
    v16bf wf = load_b_tile(W, lda, 1 * Hn + n0, k0, lane);
    v16bf wg = load_b_tile(W, lda, 2 * Hn + n0, k0, lane);
    v16bf wo = load_b_tile(W, lda, 3 * Hn + n0, k0, lane);
    ai = wmma_bf16(a, wi, ai);
    af = wmma_bf16(a, wf, af);
    ag = wmma_bf16(a, wg, ag);
    ao = wmma_bf16(a, wo, ao);
  }

  // C/D layout: elem r -> M = r + 8*(lane>=16), N = lane&15
  const int col = lane & 15;
  const int rb  = (lane >> 4) << 3;
  const int nn  = n0 + col;
  const float b_i = bias[nn],          b_f = bias[Hn + nn];
  const float b_g = bias[2 * Hn + nn], b_o = bias[3 * Hn + nn];
#pragma unroll
  for (int r = 0; r < 8; ++r) {
    const int m = m0 + rb + r;
    float iv = sigmf(ai[r] + b_i);
    float fv = sigmf(af[r] + b_f);
    float gv = tanhf(ag[r] + b_g);
    float ov = sigmf(ao[r] + b_o);
    size_t ci = (size_t)m * Hn + nn;
    float c = fv * cstate[ci] + iv * gv;
    cstate[ci] = c;
    float h = ov * tanhf(c);
    bf16_t hb = (bf16_t)h;
    hdst0[(size_t)m * ld0 + nn] = hb;
    hdst1[(size_t)m * ld1 + nn] = hb;
  }
}

// ------------------------------ head GEMMs ---------------------------------

// C[M,N] = A[M,K] @ Bmᵀ  (Bm rows are the columns of the B operand), f32 out.
__global__ __launch_bounds__(128) void gemm_bf16_kernel(
    const bf16_t* __restrict__ A, int lda,
    const bf16_t* __restrict__ Bm, int ldb,
    float* __restrict__ C, int ldc, int K) {
  const int lane = threadIdx.x & 31;
  const int wave = threadIdx.x >> 5;
  const int m0 = blockIdx.x << 4;
  const int n0 = (blockIdx.y * 4 + wave) << 4;
  v8f acc = {};
  for (int k0 = 0; k0 < K; k0 += 32) {
    v16bf a = load_a_tile(A, lda, m0, k0, lane);
    v16bf b = load_b_tile(Bm, ldb, n0, k0, lane);
    acc = wmma_bf16(a, b, acc);
  }
  const int col = lane & 15;
  const int rb  = (lane >> 4) << 3;
#pragma unroll
  for (int r = 0; r < 8; ++r)
    C[(size_t)(m0 + rb + r) * ldc + (n0 + col)] = acc[r];
}

// ----------------------- softmax / entropy / values ------------------------

// One wave per (t,b) row of logits[TB, O].
__global__ __launch_bounds__(256) void softmax_kernel(
    const float* __restrict__ logits, const int* __restrict__ actions,
    bf16_t* __restrict__ pi, float* __restrict__ dout) {
  const int lane = threadIdx.x & 31;
  const int row  = blockIdx.x * 8 + (threadIdx.x >> 5);   // t*B + b
  const float* L = logits + (size_t)row * On;
  float mx = -3.4e38f;
  for (int i = lane; i < On; i += 32) mx = fmaxf(mx, L[i]);
  mx = wave_rmax(mx);
  float s = 0.f;
  for (int i = lane; i < On; i += 32) s += __expf(L[i] - mx);
  s = wave_rsum(s);
  const float lse = __logf(s);
  float ent = 0.f;
  bf16_t* P = pi + (size_t)row * On;
  for (int i = lane; i < On; i += 32) {
    float lp = L[i] - mx - lse;
    float p  = __expf(lp);
    P[i] = (bf16_t)p;
    ent -= p * lp;
  }
  ent = wave_rsum(ent);
  const int t = row >> 8;           // / Bn
  const int b = row & (Bn - 1);
  const int act = actions[b * Tn + t];
  const float lpa = L[act] - mx - lse;
  if (lane == 0) {
    dout[(size_t)0 * TBn + b * Tn + t] = (float)act;   // actions as f32
    dout[(size_t)1 * TBn + b * Tn + t] = lpa;          // logprobs [B,T]
    dout[(size_t)2 * TBn + b * Tn + t] = ent;          // entropies [B,T]
  }
}

// values[t,b] = logits_row . s_row + pi_row . b_c   (s = pi @ W_c)
__global__ __launch_bounds__(256) void values_kernel(
    const float* __restrict__ logits, const float* __restrict__ s,
    const bf16_t* __restrict__ pi, const float* __restrict__ bc,
    float* __restrict__ dout) {
  const int lane = threadIdx.x & 31;
  const int row  = blockIdx.x * 8 + (threadIdx.x >> 5);
  const float*  L = logits + (size_t)row * On;
  const float*  S = s + (size_t)row * On;
  const bf16_t* P = pi + (size_t)row * On;
  float acc = 0.f;
  for (int i = lane; i < On; i += 32)
    acc += L[i] * S[i] + (float)P[i] * bc[i];
  acc = wave_rsum(acc);
  const int t = row >> 8;
  const int b = row & (Bn - 1);
  if (lane == 0) dout[(size_t)3 * TBn + b * Tn + t] = acc;
}

// ------------------------------- launcher ----------------------------------

extern "C" void kernel_launch(void* const* d_in, const int* in_sizes, int n_in,
                              void* d_out, int out_size, void* d_ws, size_t ws_size,
                              hipStream_t stream) {
  (void)in_sizes; (void)n_in; (void)out_size; (void)ws_size;
  // inputs (setup_inputs order); encoder_outputs (d_in[0]) is unused by the ref.
  const float* h0      = (const float*)d_in[1];
  const float* c0      = (const float*)d_in[2];
  const float* emb     = (const float*)d_in[3];
  const float* W_ih    = (const float*)d_in[4];
  const float* W_hh    = (const float*)d_in[5];
  const float* b_ih    = (const float*)d_in[6];
  const float* b_hh    = (const float*)d_in[7];
  const float* W_out   = (const float*)d_in[8];
  const float* W_c     = (const float*)d_in[9];
  const float* b_c     = (const float*)d_in[10];
  const int*   actions = (const int*)d_in[11];
  float* dout = (float*)d_out;

  // ---- workspace carve-out (256B aligned) ----
  size_t off = 0;
  char* base = (char*)d_ws;
  auto carve = [&](size_t bytes) -> void* {
    void* p = base + off;
    off += (bytes + 255) & ~(size_t)255;
    return p;
  };
  bf16_t* Wcat   = (bf16_t*)carve((size_t)Ln * 4 * Hn * 2 * Hn * 2); // 33.5 MB
  float*  bias   = (float*) carve((size_t)Ln * 4 * Hn * 4);
  bf16_t* embB   = (bf16_t*)carve((size_t)On * Hn * 2);
  bf16_t* WoutB  = (bf16_t*)carve((size_t)On * Hn * 2);
  bf16_t* WcT    = (bf16_t*)carve((size_t)On * On * 2);
  float*  cstate = (float*) carve((size_t)Ln * Bn * Hn * 4);
  bf16_t* A0[2]  = {(bf16_t*)carve((size_t)Bn * 2 * Hn * 2),
                    (bf16_t*)carve((size_t)Bn * 2 * Hn * 2)};
  bf16_t* A1[2]  = {(bf16_t*)carve((size_t)Bn * 2 * Hn * 2),
                    (bf16_t*)carve((size_t)Bn * 2 * Hn * 2)};
  bf16_t* outsB  = (bf16_t*)carve((size_t)TBn * Hn * 2);             // 67 MB
  float*  logits = (float*) carve((size_t)TBn * On * 4);             // 67 MB
  bf16_t* piB    = (bf16_t*)carve((size_t)TBn * On * 2);             // 33.5 MB
  float*  sbuf   = (float*)outsB;   // outs dead after logits GEMM; sizes match

  // ---- one-time conversions (per call; deterministic) ----
  wcat_convert_kernel<<<(Ln * 4 * Hn * 2 * Hn) / 256, 256, 0, stream>>>(W_ih, W_hh, Wcat);
  addf_kernel<<<(Ln * 4 * Hn + 255) / 256, 256, 0, stream>>>(b_ih, b_hh, bias, Ln * 4 * Hn);
  f2bf_kernel<<<(On * Hn + 255) / 256, 256, 0, stream>>>(emb, embB, On * Hn);
  f2bf_kernel<<<(On * Hn + 255) / 256, 256, 0, stream>>>(W_out, WoutB, On * Hn);
  wcT_kernel<<<(On * On + 255) / 256, 256, 0, stream>>>(W_c, WcT);
  copyf_kernel<<<(Ln * Bn * Hn + 255) / 256, 256, 0, stream>>>(c0, cstate, Ln * Bn * Hn);
  inith_kernel<<<(Ln * Bn * Hn + 255) / 256, 256, 0, stream>>>(h0, A0[0], A1[0]);

  // ---- sequential teacher-forced decode ----
  const size_t WcatL = (size_t)4 * Hn * 2 * Hn;   // per-layer Wcat stride
  dim3 lgrid(Bn / 16, Hn / 64);                   // 16 x 16 blocks, 4 waves each
  for (int t = 0; t < Tn; ++t) {
    const int cur = t & 1, nxt = cur ^ 1;
    embed_kernel<<<(Bn * Hn / 8) / 256, 256, 0, stream>>>(embB, actions, A0[cur], t);
    // layer 0: h -> A1[cur] x-region (this step L1 input) and A0[nxt] h-region
    lstm_step_kernel<<<lgrid, 128, 0, stream>>>(
        A0[cur], Wcat + 0 * WcatL, bias + 0 * 4 * Hn, cstate + 0,
        A1[cur] + 0, 2 * Hn, A0[nxt] + Hn, 2 * Hn);
    // layer 1: h -> A1[nxt] h-region (next step) and outs[t] (bf16, ld = H)
    lstm_step_kernel<<<lgrid, 128, 0, stream>>>(
        A1[cur], Wcat + 1 * WcatL, bias + 1 * 4 * Hn, cstate + (size_t)Bn * Hn,
        A1[nxt] + Hn, 2 * Hn, outsB + (size_t)t * Bn * Hn, Hn);
  }

  // ---- heads ----
  dim3 ggrid(TBn / 16, On / 64);                  // 2048 x 8
  gemm_bf16_kernel<<<ggrid, 128, 0, stream>>>(outsB, Hn, WoutB, Hn, logits, On, Hn);
  softmax_kernel<<<TBn / 8, 256, 0, stream>>>(logits, actions, piB, dout);
  gemm_bf16_kernel<<<ggrid, 128, 0, stream>>>(piB, On, WcT, On, sbuf, On, On);
  values_kernel<<<TBn / 8, 256, 0, stream>>>(logits, sbuf, piB, b_c, dout);
}
